// NNUE_37074157699726
// MI455X (gfx1250) — matrix operand order repaired and checked by reading, not compiled, not run on hardware
//
#include <hip/hip_runtime.h>

// ---------------------------------------------------------------------------
// NNUE forward for MI455X (gfx1250, wave32, WMMA + TDM).
//   h_p = clip(x_p @ W1[p]^T + b1[p], 0, 1)   p = 0,1   (4096x256 each)
//   out = [h0 h1] @ w2^T + b2                 (4096,)
// Dense f16 WMMA GEMM (x binary -> exact in f16, f32 accumulate). x tiles are
// DMA'd into LDS by the Tensor Data Mover (async, TENSORcnt), double-buffered,
// with D# padding (4 dwords / 64 dwords) to avoid LDS bank conflicts. W1 is
// converted once to f16 in d_ws (L2-resident). x streamed once from HBM —
// the real bottleneck (1.34 GB @ 23.3 TB/s ~ 58 us).
// ---------------------------------------------------------------------------

#ifndef __has_builtin
#define __has_builtin(x) 0
#endif

#if defined(__HIP_DEVICE_COMPILE__) && __has_builtin(__builtin_amdgcn_tensor_load_to_lds)
#define NN_USE_TDM 1
#else
#define NN_USE_TDM 0
#endif

typedef __attribute__((ext_vector_type(16))) _Float16 v16h;
typedef __attribute__((ext_vector_type(8)))  _Float16 v8h;
typedef __attribute__((ext_vector_type(4)))  _Float16 v4h;
typedef __attribute__((ext_vector_type(8)))  float    v8f;
typedef __attribute__((ext_vector_type(4)))  float    v4f;
typedef __attribute__((ext_vector_type(4)))  unsigned v4u;
typedef __attribute__((ext_vector_type(4)))  int      v4i;
typedef __attribute__((ext_vector_type(8)))  int      v8i;

#define NN_BATCH 4096
#define NN_IN    40960
#define NN_HID   256
#define NN_KT    32      // K per WMMA
#define NN_SKT   64      // K staged per LDS buffer step
#define NN_MT    32      // batch rows per workgroup (2 M-tiles of 16)
#define NN_SROW  68      // padded LDS row stride in floats (64 + 4 pad dwords)
#define NN_THREADS 256   // 8 wave32s

// ---- one-shot W1 fp32 -> f16 conversion into scratch ----------------------
__global__ __launch_bounds__(256) void nnue_wcvt(const float* __restrict__ w,
                                                 _Float16* __restrict__ o,
                                                 int nquads) {
  int q = blockIdx.x * blockDim.x + threadIdx.x;
  if (q < nquads) {
    const float4 f = ((const float4*)w)[q];
    v4h h;
    h[0] = (_Float16)f.x; h[1] = (_Float16)f.y;
    h[2] = (_Float16)f.z; h[3] = (_Float16)f.w;
    ((v4h*)o)[q] = h;
  }
}

#if NN_USE_TDM
// low 32 bits of the flat shared address == LDS byte offset
__device__ __forceinline__ unsigned nn_lds_off(const void* pLds) {
  return (unsigned)(unsigned long long)(uintptr_t)pLds;
}

// Issue one TDM 2D tile load: NN_MT rows x NN_SKT fp32, row stride NN_IN,
// LDS padding 4 dwords every 64 dwords (-> NN_SROW float row pitch).
__device__ __forceinline__ void nn_tdm_issue(const float* gsrc, unsigned ldsOff) {
  unsigned long long ga = (unsigned long long)(uintptr_t)gsrc;
  v4u g0;
  g0[0] = 1u;                                   // count=1, user descriptor
  g0[1] = ldsOff;                               // lds_addr (bytes)
  g0[2] = (unsigned)(ga & 0xffffffffull);       // global_addr[31:0]
  g0[3] = (unsigned)((ga >> 32) & 0x01ffffffull) | (2u << 30);  // addr[56:32], type=2
  v8i g1;
  // data_size=4B (2), pad_enable, pad_interval=64 dwords (code 5),
  // pad_amount=4 dwords (code 3)
  g1[0] = (int)((2u << 16) | (1u << 20) | (5u << 22) | (3u << 25));
  g1[1] = (int)((NN_IN & 0xffffu) << 16);                       // tensor_dim0 lo16
  g1[2] = (int)(((NN_IN >> 16) & 0xffffu) |
                ((NN_BATCH & 0xffffu) << 16));                  // dim0 hi / dim1 lo
  g1[3] = (int)(((NN_BATCH >> 16) & 0xffffu) |
                ((unsigned)NN_SKT << 16));                      // dim1 hi / tile_dim0
  g1[4] = (int)NN_MT;                                           // tile_dim1 (tile_dim2=0)
  g1[5] = (int)NN_IN;                                           // tensor_dim0_stride lo32
  g1[6] = 0;
  g1[7] = 0;
  v4i gz = {0, 0, 0, 0};
#if __clang_major__ >= 23
  v8i gz8 = {0, 0, 0, 0, 0, 0, 0, 0};
  __builtin_amdgcn_tensor_load_to_lds(g0, g1, gz, gz, gz8, 0);
#else
  __builtin_amdgcn_tensor_load_to_lds(g0, g1, gz, gz, 0);
#endif
}
#endif  // NN_USE_TDM

// Build one A fragment (16x32 f16 layout: lane-lo K0-15, lane-hi
// K16-31) from a padded fp32 LDS row, converting fp32 -> f16.
__device__ __forceinline__ v16h nn_afrag(const float* r) {
  v4f f0 = *(const v4f*)(r);
  v4f f1 = *(const v4f*)(r + 4);
  v4f f2 = *(const v4f*)(r + 8);
  v4f f3 = *(const v4f*)(r + 12);
  v16h a;
  a[0]  = (_Float16)f0[0]; a[1]  = (_Float16)f0[1];
  a[2]  = (_Float16)f0[2]; a[3]  = (_Float16)f0[3];
  a[4]  = (_Float16)f1[0]; a[5]  = (_Float16)f1[1];
  a[6]  = (_Float16)f1[2]; a[7]  = (_Float16)f1[3];
  a[8]  = (_Float16)f2[0]; a[9]  = (_Float16)f2[1];
  a[10] = (_Float16)f2[2]; a[11] = (_Float16)f2[3];
  a[12] = (_Float16)f3[0]; a[13] = (_Float16)f3[1];
  a[14] = (_Float16)f3[2]; a[15] = (_Float16)f3[3];
  return a;
}

// ---- fused NNUE kernel ----------------------------------------------------
template <bool W16>
__global__ __launch_bounds__(NN_THREADS)
void nnue_fused(const float* __restrict__ x1, const float* __restrict__ x2,
                const float* __restrict__ w32, const _Float16* __restrict__ w16,
                const float* __restrict__ b1,  const float* __restrict__ w2,
                const float* __restrict__ b2,  float* __restrict__ out) {
  __shared__ __align__(16) float sX[2][2][NN_MT * NN_SROW];  // 34 KB, dbl-buf
  __shared__ float sPart[NN_THREADS / 32][NN_MT];            // 1 KB partials

  const int tid    = threadIdx.x;
  const int wave   = tid >> 5;
  const int lane   = tid & 31;
  const int laneLo = lane & 15;
  const int hiHalf = lane >> 4;
  const int p      = wave >> 2;          // perspective (waves 0-3 / 4-7)
  const int nbase  = (wave & 3) * 64;    // this wave's 4 N-tiles of 16
  const int mbase  = blockIdx.x * NN_MT;
  const int k0a    = hiHalf * 16;        // A fragment K offset (contiguous)
  const int k0b    = hiHalf * 16;        // B fragment K offset (contiguous)

#if NN_USE_TDM
  const float* xp = p ? x2 : x1;         // TDM issued by waves 0 and 4
#else
  const int lidx = tid & 127;
  const int lp   = tid >> 7;
  const int lrow = lidx >> 2;
  const int lseg = (lidx & 3) * 16;
  const float* xrow = (lp ? x2 : x1) + (size_t)(mbase + lrow) * NN_IN + lseg;
#endif

  const _Float16* wp16[4];
  const float*    wp32[4];
#pragma unroll
  for (int nt = 0; nt < 4; ++nt) {
    const size_t hrow = (size_t)p * NN_HID + nbase + nt * 16 + laneLo;
    if constexpr (W16) wp16[nt] = w16 + hrow * NN_IN + k0b;
    else               wp32[nt] = w32 + hrow * NN_IN + k0b;
  }

  v8f acc[2][4];
  const v8f vz = {0.f, 0.f, 0.f, 0.f, 0.f, 0.f, 0.f, 0.f};
#pragma unroll
  for (int mt = 0; mt < 2; ++mt)
#pragma unroll
    for (int nt = 0; nt < 4; ++nt) acc[mt][nt] = vz;

  // ---- prologue: stage ks=0 into buffer 0 ----
#if NN_USE_TDM
  if ((wave & 3) == 0) {
    nn_tdm_issue(xp + (size_t)mbase * NN_IN, nn_lds_off(&sX[0][p][0]));
    __builtin_amdgcn_s_wait_tensorcnt((short)0);
  }
#else
#pragma unroll
  for (int j = 0; j < 4; ++j) {
    float4 f = *(const float4*)(xrow + j * 4);
    *(float4*)&sX[0][lp][lrow * NN_SROW + lseg + j * 4] = f;
  }
#endif
  __syncthreads();

  for (int ks = 0; ks < NN_IN; ks += NN_SKT) {
    const int  cur  = (ks / NN_SKT) & 1;
    const bool more = (ks + NN_SKT) < NN_IN;

    // kick off the next x tile (fully async behind the WMMA work)
#if NN_USE_TDM
    if (more && (wave & 3) == 0)
      nn_tdm_issue(xp + (size_t)mbase * NN_IN + ks + NN_SKT,
                   nn_lds_off(&sX[cur ^ 1][p][0]));
#else
    float4 stg[4];
    if (more) {
#pragma unroll
      for (int j = 0; j < 4; ++j)
        stg[j] = *(const float4*)(xrow + ks + NN_SKT + j * 4);
    }
#endif

    if (ks + 4 * NN_SKT < NN_IN) {
      if constexpr (W16) __builtin_prefetch(wp16[0] + ks + 4 * NN_SKT, 0, 0);
      else               __builtin_prefetch(wp32[0] + ks + 4 * NN_SKT, 0, 0);
    }

#pragma unroll
    for (int kk = 0; kk < NN_SKT; kk += NN_KT) {
      // A fragments from padded fp32 LDS (convert to f16 in-register)
      v16h af[2];
#pragma unroll
      for (int mt = 0; mt < 2; ++mt)
        af[mt] = nn_afrag(&sX[cur][p][(mt * 16 + laneLo) * NN_SROW + kk + k0a]);

      // all 4 B fragments in distinct registers: one load clause, one wait
      v16h bf[4];
#pragma unroll
      for (int nt = 0; nt < 4; ++nt) {
        if constexpr (W16) {
          bf[nt] = *(const v16h*)(wp16[nt] + ks + kk);  // 32B -> 2x b128
        } else {
          const float* q = wp32[nt] + ks + kk;
#pragma unroll
          for (int i = 0; i < 16; ++i) bf[nt][i] = (_Float16)q[i];
        }
      }

#pragma unroll
      for (int nt = 0; nt < 4; ++nt) {
        acc[0][nt] = __builtin_amdgcn_wmma_f32_16x16x32_f16(
            false, af[0], false, bf[nt], (short)0, acc[0][nt], false, false);
        acc[1][nt] = __builtin_amdgcn_wmma_f32_16x16x32_f16(
            false, af[1], false, bf[nt], (short)0, acc[1][nt], false, false);
      }
    }

#if NN_USE_TDM
    if (more && (wave & 3) == 0)
      __builtin_amdgcn_s_wait_tensorcnt((short)0);
#else
    if (more) {
#pragma unroll
      for (int j = 0; j < 4; ++j)
        *(float4*)&sX[cur ^ 1][lp][lrow * NN_SROW + lseg + j * 4] = stg[j];
    }
#endif
    __syncthreads();
  }

  // ---- fused epilogue: bias + clip[0,1] + layer-2 dot, deterministic ----
  float biasv[4], w2v[4];
#pragma unroll
  for (int nt = 0; nt < 4; ++nt) {
    const int h = p * NN_HID + nbase + nt * 16 + laneLo;
    biasv[nt] = b1[h];
    w2v[nt]   = w2[h];
  }
#pragma unroll
  for (int mt = 0; mt < 2; ++mt) {
    float rs[8];
#pragma unroll
    for (int j = 0; j < 8; ++j) rs[j] = 0.f;
#pragma unroll
    for (int nt = 0; nt < 4; ++nt)
#pragma unroll
      for (int j = 0; j < 8; ++j) {
        float v = acc[mt][nt][j] + biasv[nt];
        v = fminf(fmaxf(v, 0.f), 1.f);
        rs[j] += v * w2v[nt];
      }
    // reduce across the 16 N-lanes of each half (xor stays within halves)
#pragma unroll
    for (int off = 1; off < 16; off <<= 1)
#pragma unroll
      for (int j = 0; j < 8; ++j) rs[j] += __shfl_xor(rs[j], off, 32);
    if (laneLo == 0) {
#pragma unroll
      for (int j = 0; j < 8; ++j)
        sPart[wave][mt * 16 + hiHalf * 8 + j] = rs[j];
    }
  }
  __syncthreads();
  if (tid < NN_MT) {
    float s = 0.f;
#pragma unroll
    for (int w = 0; w < NN_THREADS / 32; ++w) s += sPart[w][tid];
    out[mbase + tid] = s + b2[0];
  }
}

// ---------------------------------------------------------------------------
extern "C" void kernel_launch(void* const* d_in, const int* in_sizes, int n_in,
                              void* d_out, int out_size, void* d_ws,
                              size_t ws_size, hipStream_t stream) {
  (void)in_sizes; (void)n_in; (void)out_size;
  const float* x1 = (const float*)d_in[0];
  const float* x2 = (const float*)d_in[1];
  const float* w1 = (const float*)d_in[2];
  const float* b1 = (const float*)d_in[3];
  const float* w2 = (const float*)d_in[4];
  const float* b2 = (const float*)d_in[5];
  float* out = (float*)d_out;

  const size_t wElems = (size_t)2 * NN_HID * NN_IN;
  const dim3 grid(NN_BATCH / NN_MT), block(NN_THREADS);

  if (ws_size >= wElems * sizeof(_Float16)) {
    _Float16* w16 = (_Float16*)d_ws;
    const int nquads = (int)(wElems / 4);
    nnue_wcvt<<<(nquads + 255) / 256, 256, 0, stream>>>(w1, w16, nquads);
    nnue_fused<true><<<grid, block, 0, stream>>>(x1, x2, w1, w16, b1, w2, b2, out);
  } else {
    nnue_fused<false><<<grid, block, 0, stream>>>(x1, x2, w1,
                                                  (const _Float16*)nullptr,
                                                  b1, w2, b2, out);
  }
}
